// SimpleAttention_32452772888747
// MI455X (gfx1250) — matrix operand order repaired and checked by reading, not compiled, not run on hardware
//
#include <hip/hip_runtime.h>
#include <hip/hip_bf16.h>
#include <stdint.h>

// ---------------------------------------------------------------------------
// Galerkin attention on MI455X (gfx1250): bf16 WMMA pipeline, fp32 accumulate.
// B=4, S=8192, D=1024, H=8, DK=128, P=1
// ---------------------------------------------------------------------------

#define BB 4
#define SS 8192
#define DD 1024
#define HH 8
#define DKK 128
#define MM (BB * SS)          // 32768
#define KX 1056               // padded H*(DK+P)=1032 -> multiple of 32

typedef __attribute__((ext_vector_type(16))) __bf16 v16bf;
typedef __attribute__((ext_vector_type(8)))  float  v8f;
typedef unsigned int u32x4 __attribute__((ext_vector_type(4)));
typedef int          i32x8 __attribute__((ext_vector_type(8)));
typedef int          i32x4 __attribute__((ext_vector_type(4)));

#ifndef __has_builtin
#define __has_builtin(x) 0
#endif
#if __has_builtin(__builtin_amdgcn_tensor_load_to_lds) && \
    __has_builtin(__builtin_amdgcn_s_wait_tensorcnt)
#define USE_TDM 1
#else
#define USE_TDM 0
#endif

union V16 {
    v16bf  v;
    __bf16 e[16];
    uint4  q[2];
};

// Load 16 bf16 elements (two 16B chunks) into a WMMA operand.
__device__ __forceinline__ v16bf load16_bf(const __bf16* p0, const __bf16* p1) {
    V16 u;
    u.q[0] = *(const uint4*)p0;
    u.q[1] = *(const uint4*)p1;
    return u.v;
}

__device__ __forceinline__ v8f wmma_bf16(v16bf a, v16bf b, v8f c) {
    return __builtin_amdgcn_wmma_f32_16x16x32_bf16(
        /*neg_a=*/false, a, /*neg_b=*/false, b,
        /*c_mod=*/(short)0, c, /*reuse_a=*/false, /*reuse_b=*/false);
}

#if USE_TDM
// Tensor Data Mover: load a 2D tile (tile_d1 rows x tile_d0 elems of 2-byte
// data, row stride stride0 elems) from global into LDS at lds_off.
// D# packing per CDNA5 ISA (count=1, type=2).
// This toolchain exposes the 6-arg builtin (extra int32x8 before cpol).
__device__ __forceinline__ void tdm_load_tile_2d_bf16(
    unsigned lds_off, const void* gaddr, unsigned tensor_d0, unsigned tensor_d1,
    unsigned tile_d0, unsigned tile_d1, unsigned stride0) {
    unsigned long long ga = (unsigned long long)(uintptr_t)gaddr;
    u32x4 g0;
    g0[0] = 1u;                                    // count=1, user-mode D#
    g0[1] = lds_off;                               // lds_addr (bytes)
    g0[2] = (unsigned)(ga & 0xFFFFFFFFu);          // global_addr[31:0]
    g0[3] = (unsigned)((ga >> 32) & 0x01FFFFFFu)   // global_addr[56:32]
            | 0x80000000u;                         // type=2 ("image")
    i32x8 g1;
    g1[0] = (int)(1u << 16);                       // data_size=1 -> 2 bytes
    g1[1] = (int)((tensor_d0 & 0xFFFFu) << 16);    // tensor_dim0[15:0]
    g1[2] = (int)((tensor_d0 >> 16) | ((tensor_d1 & 0xFFFFu) << 16));
    g1[3] = (int)((tensor_d1 >> 16) | (tile_d0 << 16));  // tile_dim0
    g1[4] = (int)tile_d1;                          // tile_dim1 (tile_dim2=0)
    g1[5] = (int)stride0;                          // tensor_dim0_stride[31:0]
    g1[6] = 0;
    g1[7] = 0;
    i32x4 z4 = {0, 0, 0, 0};                       // groups 2/3 unused (2D)
    i32x8 z8 = {0, 0, 0, 0, 0, 0, 0, 0};
    __builtin_amdgcn_tensor_load_to_lds(g0, g1, z4, z4, z8, 0);
}
#endif

// ---------------------------------------------------------------------------
// Elementwise fp32 -> bf16 (vectorized x4)
// ---------------------------------------------------------------------------
__global__ void cvt_bf16x4_kernel(const float* __restrict__ src,
                                  __bf16* __restrict__ dst, int n4) {
    int i = blockIdx.x * blockDim.x + threadIdx.x;
    if (i >= n4) return;
    float4 f = ((const float4*)src)[i];
    union { __bf16 e[4]; uint2 u; } o;
    o.e[0] = (__bf16)f.x; o.e[1] = (__bf16)f.y;
    o.e[2] = (__bf16)f.z; o.e[3] = (__bf16)f.w;
    ((uint2*)dst)[i] = o.u;
}

// Wfc: (1024 x 1032) fp32 -> (1024 x 1056) bf16 row-padded with zeros
__global__ void cvt_wfc_kernel(const float* __restrict__ src,
                               __bf16* __restrict__ dst) {
    int i = blockIdx.x * blockDim.x + threadIdx.x;  // over 1024*1056
    if (i >= DD * KX) return;
    int r = i / KX, c = i - r * KX;
    dst[i] = (c < DD + HH) ? (__bf16)src[r * (DD + HH) + c] : (__bf16)0.0f;
}

// Zero pad columns [1032,1056) of the x buffer (avoid poison NaNs in WMMA)
__global__ void zero_pad_kernel(__bf16* __restrict__ x) {
    int i = blockIdx.x * blockDim.x + threadIdx.x;  // over MM*24
    if (i >= MM * (KX - DD - HH)) return;
    int row = i / (KX - DD - HH), c = i - row * (KX - DD - HH);
    x[(size_t)row * KX + (DD + HH) + c] = (__bf16)0.0f;
}

// ---------------------------------------------------------------------------
// GEMM: out[m,n] = sum_k A[m,k] * W[n,k] + bias[n]
//   A: bf16 row-major (lda elems); W: bf16 row-major N x K (= B col-major)
// Block tile 128x128, 8 waves, each wave 32(M) x 64(N) = 2x4 WMMA tiles.
// Pure b128 loads + v_wmma inner loop; next-tile global_prefetch.
// ---------------------------------------------------------------------------
template <int OUTBF16>
__global__ void __launch_bounds__(256)
gemm_xwT_kernel(const __bf16* __restrict__ A, const __bf16* __restrict__ W,
                const float* __restrict__ bias, void* __restrict__ outv,
                int K, int lda, int ldw, int ldo) {
    const int tid  = threadIdx.x;
    const int lane = tid & 31;
    const int wid  = tid >> 5;
    const int r0   = lane & 15;
    const int hi   = lane >> 4;

    const int nblk = blockIdx.x * 128;
    const int mblk = blockIdx.y * 128;
    const int wm   = (wid & 3) * 32;
    const int wn   = (wid >> 2) * 64;

    v8f c[2][4];
#pragma unroll
    for (int i = 0; i < 2; ++i)
#pragma unroll
        for (int j = 0; j < 4; ++j) c[i][j] = (v8f)0.0f;

    const __bf16* a0p = A + (size_t)(mblk + wm + r0) * lda + hi * 8;
    const __bf16* b0p = W + (size_t)(nblk + wn + r0) * ldw + hi * 8;

    for (int k0 = 0; k0 < K; k0 += 32) {
        if (k0 + 32 < K) {  // speculative prefetch of next k-tile
            __builtin_prefetch(a0p + k0 + 32, 0, 0);
            __builtin_prefetch(b0p + k0 + 32, 0, 0);
        }
        v16bf a[2];
#pragma unroll
        for (int mt = 0; mt < 2; ++mt) {
            const __bf16* ap = a0p + (size_t)(mt * 16) * lda + k0;
            a[mt] = load16_bf(ap, ap + 16);
        }
        v16bf b[4];
#pragma unroll
        for (int nt = 0; nt < 4; ++nt) {
            const __bf16* bp = b0p + (size_t)(nt * 16) * ldw + k0;
            b[nt] = load16_bf(bp, bp + 16);
        }
#pragma unroll
        for (int mt = 0; mt < 2; ++mt)
#pragma unroll
            for (int nt = 0; nt < 4; ++nt)
                c[mt][nt] = wmma_bf16(a[mt], b[nt], c[mt][nt]);
    }

#pragma unroll
    for (int mt = 0; mt < 2; ++mt) {
#pragma unroll
        for (int nt = 0; nt < 4; ++nt) {
            const int col = nblk + wn + nt * 16 + r0;
            const float bv = bias ? bias[col] : 0.0f;
#pragma unroll
            for (int r = 0; r < 8; ++r) {
                const size_t row = (size_t)(mblk + wm + mt * 16 + r + hi * 8);
                const float val = c[mt][nt][r] + bv;
                if (OUTBF16)
                    ((__bf16*)outv)[row * ldo + col] = (__bf16)val;
                else
                    ((float*)outv)[row * ldo + col] = val;
            }
        }
    }
}

// ---------------------------------------------------------------------------
// Per-head LayerNorm over DK=128: one wave per (b,s,h) row. bf16 in/out,
// fp32 statistics.
// ---------------------------------------------------------------------------
__global__ void __launch_bounds__(256)
ln_head_kernel(const __bf16* __restrict__ x, const float* __restrict__ g,
               const float* __restrict__ beta, __bf16* __restrict__ y) {
    const int wrow = blockIdx.x * 8 + (threadIdx.x >> 5);  // over B*S*H
    const int lane = threadIdx.x & 31;
    const int h = wrow & (HH - 1);
    const size_t bs = (size_t)(wrow >> 3);

    const __bf16* xp = x + bs * DD + h * DKK;
    float vals[4], s = 0.0f, sq = 0.0f;
#pragma unroll
    for (int i = 0; i < 4; ++i) {
        float t = (float)xp[lane + 32 * i];
        vals[i] = t;
        s += t;
        sq += t * t;
    }
#pragma unroll
    for (int off = 16; off > 0; off >>= 1) {
        s  += __shfl_xor(s, off, 32);
        sq += __shfl_xor(sq, off, 32);
    }
    const float m = s * (1.0f / DKK);
    const float var = sq * (1.0f / DKK) - m * m;
    const float rstd = rsqrtf(var + 1e-5f);
#pragma unroll
    for (int i = 0; i < 4; ++i) {
        const int d = lane + 32 * i;
        const float yy = g[h * DKK + d] * (vals[i] - m) * rstd + beta[h * DKK + d];
        y[bs * DD + h * DKK + d] = (__bf16)yy;
    }
}

// ---------------------------------------------------------------------------
// p_attn core: per (b,h) compute Kt@V (128x128) over S via WMMA.
// Tile staging: TDM (tensor_load_to_lds) into row-major LDS, then in-LDS
// transpose; fallback = direct global->LDS transposed staging.
// Also rank-1 pos sums A0e = pTv/S, Ad0 = kTp/S, A00 = pTp/S.
// pattnT stored e-major so the next GEMM's B operand is layout-free.
// ---------------------------------------------------------------------------
__global__ void __launch_bounds__(256)
ktv_kernel(const __bf16* __restrict__ kln, const __bf16* __restrict__ vln,
           const float* __restrict__ pos, __bf16* __restrict__ pattnT,
           float* __restrict__ A0e, float* __restrict__ Ad0,
           float* __restrict__ A00) {
    const int bh = blockIdx.x;            // b*H + h
    const int b = bh >> 3, h = bh & 7;

#if USE_TDM
    __shared__ __bf16 kRow[32][128];      // row-major TDM landing tiles
    __shared__ __bf16 vRow[32][128];
#endif
    __shared__ __bf16 kT[128][40];        // [d][s], padded stride
    __shared__ __bf16 vT[128][40];        // [e][s]
    __shared__ float  ppos[32];

    const int tid  = threadIdx.x;
    const int lane = tid & 31;
    const int wid  = tid >> 5;
    const int r0   = lane & 15;
    const int hi   = lane >> 4;

    v8f c[8];
#pragma unroll
    for (int i = 0; i < 8; ++i) c[i] = (v8f)0.0f;
    float ad0_acc = 0.0f, a0e_acc = 0.0f, a00_acc = 0.0f;

    for (int s0 = 0; s0 < SS; s0 += 32) {
        const size_t gbase = ((size_t)(b * SS + s0)) * DD + h * DKK;
#if USE_TDM
        if (wid == 0) {
            tdm_load_tile_2d_bf16((unsigned)(uintptr_t)&kRow[0][0], kln + gbase,
                                  DKK, 32, DKK, 32, DD);
            tdm_load_tile_2d_bf16((unsigned)(uintptr_t)&vRow[0][0], vln + gbase,
                                  DKK, 32, DKK, 32, DD);
            __builtin_amdgcn_s_wait_tensorcnt(0);
        }
        if (tid < 32) ppos[tid] = pos[(size_t)b * SS + s0 + tid];
        __syncthreads();
        {   // in-LDS transpose kRow/vRow -> kT/vT
            const int si = tid >> 3;            // 0..31
            const int dg = (tid & 7) * 16;      // 0,16,..,112
            V16 tk, tv;
            tk.q[0] = *(const uint4*)&kRow[si][dg];
            tk.q[1] = *(const uint4*)&kRow[si][dg + 8];
            tv.q[0] = *(const uint4*)&vRow[si][dg];
            tv.q[1] = *(const uint4*)&vRow[si][dg + 8];
#pragma unroll
            for (int j = 0; j < 16; ++j) {
                kT[dg + j][si] = tk.e[j];
                vT[dg + j][si] = tv.e[j];
            }
        }
        __syncthreads();
#else
        {   // direct global -> transposed LDS
            const int si = tid >> 3;
            const int dg = (tid & 7) * 16;
            const __bf16* kp = kln + gbase + (size_t)si * DD + dg;
            const __bf16* vp = vln + gbase + (size_t)si * DD + dg;
            V16 tk, tv;
            tk.q[0] = *(const uint4*)kp;  tk.q[1] = *(const uint4*)(kp + 8);
            tv.q[0] = *(const uint4*)vp;  tv.q[1] = *(const uint4*)(vp + 8);
#pragma unroll
            for (int j = 0; j < 16; ++j) {
                kT[dg + j][si] = tk.e[j];
                vT[dg + j][si] = tv.e[j];
            }
            if (tid < 32) ppos[tid] = pos[(size_t)b * SS + s0 + tid];
        }
        __syncthreads();
#endif

        // ---- WMMA: wave wid owns d-tile wid (16 rows of d), all 8 e-tiles ----
        {
            const __bf16* ap = &kT[wid * 16 + r0][hi * 8];
            v16bf a = load16_bf(ap, ap + 16);   // second chunk: s += 16
#pragma unroll
            for (int et = 0; et < 8; ++et) {
                const __bf16* bp = &vT[et * 16 + r0][hi * 8];
                v16bf bb = load16_bf(bp, bp + 16);
                c[et] = wmma_bf16(a, bb, c[et]);
            }
        }

        // ---- rank-1 pos sums (waves 0-3: per-d / per-e dot with pos) ----
        if (tid < 128) {
            float aa = 0.0f, bb2 = 0.0f;
#pragma unroll 8
            for (int s = 0; s < 32; ++s) {
                const float p = ppos[s];
                aa  += p * (float)kT[tid][s];
                bb2 += p * (float)vT[tid][s];
            }
            ad0_acc += aa;
            a0e_acc += bb2;
        }
        if (wid == 7) {
            const float p = ppos[lane];
            a00_acc += p * p;
        }
        __syncthreads();
    }

    const float invS = 1.0f / (float)SS;

    // store transposed: pattnT[bh][e][d]
#pragma unroll
    for (int et = 0; et < 8; ++et) {
        const int e = et * 16 + r0;
#pragma unroll
        for (int r = 0; r < 8; ++r) {
            const int d = wid * 16 + r + hi * 8;
            pattnT[((size_t)bh * 128 + e) * 128 + d] = (__bf16)(c[et][r] * invS);
        }
    }
    if (tid < 128) {
        Ad0[bh * 128 + tid] = ad0_acc * invS;
        A0e[bh * 128 + tid] = a0e_acc * invS;
    }
    if (wid == 7) {
#pragma unroll
        for (int off = 16; off > 0; off >>= 1) a00_acc += __shfl_xor(a00_acc, off, 32);
        if (lane == 0) A00[bh] = a00_acc * invS;
    }
}

// ---------------------------------------------------------------------------
// x[:,0] per head: x0 = pos*A00 + q . Ad0  (one thread per (b,s,h))
// ---------------------------------------------------------------------------
__global__ void x0_kernel(const __bf16* __restrict__ qbf,
                          const float* __restrict__ pos,
                          const float* __restrict__ Ad0,
                          const float* __restrict__ A00,
                          __bf16* __restrict__ xfull) {
    const int idx = blockIdx.x * blockDim.x + threadIdx.x;
    if (idx >= MM * HH) return;
    const int h = idx & (HH - 1);
    const int bs = idx >> 3;
    const int b = bs >> 13;
    const int bh = b * HH + h;
    const __bf16* qp = qbf + (size_t)bs * DD + h * DKK;
    const float* ad = Ad0 + bh * DKK;
    float acc = 0.0f;
#pragma unroll 8
    for (int d = 0; d < DKK; ++d) acc += (float)qp[d] * ad[d];
    const float x0 = pos[bs] * A00[bh] + acc;
    xfull[(size_t)bs * KX + h * (DKK + 1)] = (__bf16)x0;
}

// ---------------------------------------------------------------------------
// x[:,1+e] = q @ Ade + pos * A0e[e]    (per (b, s-block of 128, h))
// A = q (row-major bf16), B = pattnT (e-major, d contiguous) -> both direct.
// ---------------------------------------------------------------------------
__global__ void __launch_bounds__(256)
qpa_kernel(const __bf16* __restrict__ qbf, const __bf16* __restrict__ pattnT,
           const float* __restrict__ A0e, const float* __restrict__ pos,
           __bf16* __restrict__ xfull) {
    const int sb = blockIdx.x * 128;
    const int b  = blockIdx.y;
    const int h  = blockIdx.z;
    const int bh = b * HH + h;

    const int tid  = threadIdx.x;
    const int lane = tid & 31;
    const int wid  = tid >> 5;
    const int r0   = lane & 15;
    const int hi   = lane >> 4;

    v8f c[8];
#pragma unroll
    for (int i = 0; i < 8; ++i) c[i] = (v8f)0.0f;

#pragma unroll
    for (int k0 = 0; k0 < DKK; k0 += 32) {
        const size_t arow = (size_t)b * SS + sb + wid * 16 + r0;
        const __bf16* ap = qbf + arow * DD + h * DKK + k0 + hi * 8;
        v16bf a = load16_bf(ap, ap + 16);
#pragma unroll
        for (int et = 0; et < 8; ++et) {
            const __bf16* bp =
                pattnT + ((size_t)bh * 128 + et * 16 + r0) * 128 + k0 + hi * 8;
            v16bf bb = load16_bf(bp, bp + 16);
            c[et] = wmma_bf16(a, bb, c[et]);
        }
    }

#pragma unroll
    for (int et = 0; et < 8; ++et) {
        const int e = et * 16 + r0;
        const float a0 = A0e[bh * 128 + e];
#pragma unroll
        for (int r = 0; r < 8; ++r) {
            const int srow = sb + wid * 16 + r + hi * 8;
            const float p = pos[(size_t)b * SS + srow];
            const float val = c[et][r] + p * a0;
            xfull[((size_t)b * SS + srow) * KX + h * (DKK + 1) + 1 + e] = (__bf16)val;
        }
    }
}

// ---------------------------------------------------------------------------
// host-side launch
// ---------------------------------------------------------------------------
extern "C" void kernel_launch(void* const* d_in, const int* in_sizes, int n_in,
                              void* d_out, int out_size, void* d_ws, size_t ws_size,
                              hipStream_t stream) {
    const float* query = (const float*)d_in[0];
    const float* key   = (const float*)d_in[1];
    const float* value = (const float*)d_in[2];
    const float* pos   = (const float*)d_in[3];
    const float* Wq    = (const float*)d_in[4];
    const float* bq    = (const float*)d_in[5];
    const float* Wk    = (const float*)d_in[6];
    const float* bk    = (const float*)d_in[7];
    const float* Wv    = (const float*)d_in[8];
    const float* bv    = (const float*)d_in[9];
    const float* gK    = (const float*)d_in[10];
    const float* betaK = (const float*)d_in[11];
    const float* gV    = (const float*)d_in[12];
    const float* betaV = (const float*)d_in[13];
    const float* Wfc   = (const float*)d_in[14];
    const float* bfc   = (const float*)d_in[15];

    char* p = (char*)d_ws;
    auto carve = [&](size_t bytes) -> void* {
        void* r = (void*)p;
        p += (bytes + 255) & ~(size_t)255;
        return r;
    };
    __bf16* Wq_bf  = (__bf16*)carve((size_t)DD * DD * 2);
    __bf16* Wk_bf  = (__bf16*)carve((size_t)DD * DD * 2);
    __bf16* Wv_bf  = (__bf16*)carve((size_t)DD * DD * 2);
    __bf16* Wfc_bf = (__bf16*)carve((size_t)DD * KX * 2);
    __bf16* qx_bf  = (__bf16*)carve((size_t)MM * DD * 2);   // bf16 inputs
    __bf16* kx_bf  = (__bf16*)carve((size_t)MM * DD * 2);
    __bf16* vx_bf  = (__bf16*)carve((size_t)MM * DD * 2);
    __bf16* q_bf   = (__bf16*)carve((size_t)MM * DD * 2);   // projections
    __bf16* k_bf   = (__bf16*)carve((size_t)MM * DD * 2);
    __bf16* v_bf   = (__bf16*)carve((size_t)MM * DD * 2);
    __bf16* kln    = (__bf16*)carve((size_t)MM * DD * 2);
    __bf16* vln    = (__bf16*)carve((size_t)MM * DD * 2);
    __bf16* pattnT = (__bf16*)carve((size_t)BB * HH * 128 * 128 * 2);
    float*  A0e    = (float*) carve((size_t)BB * HH * 128 * 4);
    float*  Ad0    = (float*) carve((size_t)BB * HH * 128 * 4);
    float*  A00    = (float*) carve((size_t)BB * HH * 4);
    __bf16* xfull  = (__bf16*)carve((size_t)MM * KX * 2);

    // 1) weights + activations -> bf16
    {
        const int nw4 = DD * DD / 4;
        cvt_bf16x4_kernel<<<(nw4 + 255) / 256, 256, 0, stream>>>(Wq, Wq_bf, nw4);
        cvt_bf16x4_kernel<<<(nw4 + 255) / 256, 256, 0, stream>>>(Wk, Wk_bf, nw4);
        cvt_bf16x4_kernel<<<(nw4 + 255) / 256, 256, 0, stream>>>(Wv, Wv_bf, nw4);
        cvt_wfc_kernel<<<(DD * KX + 255) / 256, 256, 0, stream>>>(Wfc, Wfc_bf);
        const int na4 = MM * DD / 4;
        cvt_bf16x4_kernel<<<(na4 + 255) / 256, 256, 0, stream>>>(query, qx_bf, na4);
        cvt_bf16x4_kernel<<<(na4 + 255) / 256, 256, 0, stream>>>(key,   kx_bf, na4);
        cvt_bf16x4_kernel<<<(na4 + 255) / 256, 256, 0, stream>>>(value, vx_bf, na4);
    }

    // 2) QKV projections (pure bf16 WMMA GEMM)
    {
        dim3 grid(DD / 128, MM / 128);
        gemm_xwT_kernel<1><<<grid, 256, 0, stream>>>(qx_bf, Wq_bf, bq, q_bf,
                                                     DD, DD, DD, DD);
        gemm_xwT_kernel<1><<<grid, 256, 0, stream>>>(kx_bf, Wk_bf, bk, k_bf,
                                                     DD, DD, DD, DD);
        gemm_xwT_kernel<1><<<grid, 256, 0, stream>>>(vx_bf, Wv_bf, bv, v_bf,
                                                     DD, DD, DD, DD);
    }

    // 3) per-head LayerNorm of K,V
    {
        const int rows = MM * HH;
        ln_head_kernel<<<rows / 8, 256, 0, stream>>>(k_bf, gK, betaK, kln);
        ln_head_kernel<<<rows / 8, 256, 0, stream>>>(v_bf, gV, betaV, vln);
    }

    // 4) KtV + pos rank-1 sums (TDM-staged)
    ktv_kernel<<<BB * HH, 256, 0, stream>>>(kln, vln, pos, pattnT, A0e, Ad0, A00);

    // 5) build x (padded width 1056)
    {
        const int npad = MM * (KX - DD - HH);
        zero_pad_kernel<<<(npad + 255) / 256, 256, 0, stream>>>(xfull);
        x0_kernel<<<(MM * HH + 255) / 256, 256, 0, stream>>>(q_bf, pos, Ad0, A00,
                                                             xfull);
        dim3 grid(SS / 128, BB, HH);
        qpa_kernel<<<grid, 256, 0, stream>>>(q_bf, pattnT, A0e, pos, xfull);
    }

    // 6) output FC: out = x @ Wfc.T + bfc (fp32 out)
    {
        dim3 grid(DD / 128, MM / 128);
        gemm_xwT_kernel<0><<<grid, 256, 0, stream>>>(xfull, Wfc_bf, bfc,
                                                     d_out, KX, KX, KX, DD);
    }
}
